// QuantumAttention_42073499632262
// MI455X (gfx1250) — matrix operand order repaired and checked by reading, not compiled, not run on hardware
//
#include <hip/hip_runtime.h>
#include <cstdint>
#include <cstddef>

// ============================================================================
// QuantumAttention for MI455X (gfx1250), wave32 + WMMA bf16 + async-LDS copies.
//
// B=4, S=1024, E=1024, H=16, HD=64.
//
// Roofline: the mandatory post-mix probs output is 268MB; this version touches
// it exactly twice (one write in the fused probs+mix pass, one f32 read in the
// ctx GEMM) = 536MB ~ 23us at 23.3TB/s, instead of 4 passes. Scores are
// recomputed from a tiny (512KB) max/1-over-sum stats table; the 16x16 head
// mix is done element-wise in WMMA C-fragment registers (same (lane,vgpr) ->
// same (q,k) for every head). Hadamard + cos(phase) are folded into the Q/K
// projection weights. All matmuls are v_wmma_f32_16x16x32_bf16 with f32 accum.
// ============================================================================

typedef __bf16 bf16;
typedef __attribute__((ext_vector_type(16))) __bf16 v16bf;
typedef __attribute__((ext_vector_type(8)))  __bf16 v8bf;
typedef __attribute__((ext_vector_type(8)))  float  v8f;

union BF16x16 { v16bf v; v8bf h[2]; };

__device__ __forceinline__ bf16 f2bf(float f) {
  unsigned u = __float_as_uint(f);
  u += 0x7FFFu + ((u >> 16) & 1u);               // round-to-nearest-even
  unsigned short s = (unsigned short)(u >> 16);
  return __builtin_bit_cast(bf16, s);
}

#define EE 1024
#define SS 1024
#define HH 16
#define DD 64

// ---- CDNA5 async global->LDS copy (ASYNCcnt path), per ISA 15.18.3 ----------
#define USE_ASYNC_LDS 1

__device__ __forceinline__ void async_g2l_b128(void* lds_dst, const void* gsrc) {
  // Generic LDS pointer: addr[31:0] is the LDS byte offset (ISA 10.2).
  uint32_t loff = (uint32_t)(uintptr_t)lds_dst;
  asm volatile("global_load_async_to_lds_b128 %0, %1, off"
               :: "v"(loff), "v"(gsrc) : "memory");
}
__device__ __forceinline__ void wait_async0() {
  asm volatile("s_wait_asynccnt 0" ::: "memory");
}

// ----------------------------------------------------------------------------
// Kernel 0: fold Hadamard + cos(phase) into Q/K weights; convert Wv/Wo to bf16.
// Effective B-matrix layout is [n][k] so GEMM B-tiles stage contiguously.
// ----------------------------------------------------------------------------
__global__ __launch_bounds__(256) void prep_weights(
    const float* __restrict__ Wq, const float* __restrict__ Wk,
    const float* __restrict__ Wv, const float* __restrict__ Wo,
    const float* __restrict__ phase, const float* __restrict__ had,
    bf16* __restrict__ WqE, bf16* __restrict__ WkE,
    bf16* __restrict__ WvT, bf16* __restrict__ WoT)
{
  int idx = blockIdx.x * 256 + threadIdx.x;       // over E*E
  int n = idx >> 10, k = idx & 1023;
  WvT[idx] = f2bf(Wv[idx]);                       // B[k][n]=Wv[n,k]; [n][k] = copy
  WoT[idx] = f2bf(Wo[idx]);
  int h = n >> 6, d = n & 63;
  float cp = __cosf(phase[n]);                    // phase flat [H,HD] == n
  float sq = 0.f, sk = 0.f;
  #pragma unroll 8
  for (int c = 0; c < 64; ++c) {
    float hv = had[(c << 6) + d];
    sq = fmaf(Wq[((h << 6) + c) * EE + k], hv, sq);
    sk = fmaf(Wk[((h << 6) + c) * EE + k], hv, sk);
  }
  WqE[idx] = f2bf(sq * cp);
  WkE[idx] = f2bf(sk * cp);
}

__global__ __launch_bounds__(256) void prep_bias(
    const float* __restrict__ bq, const float* __restrict__ bk,
    const float* __restrict__ phase, const float* __restrict__ had,
    float* __restrict__ bqE, float* __restrict__ bkE)
{
  int n = blockIdx.x * 256 + threadIdx.x;         // over E
  int h = n >> 6, d = n & 63;
  float cp = __cosf(phase[n]);
  float sq = 0.f, sk = 0.f;
  for (int c = 0; c < 64; ++c) {
    float hv = had[(c << 6) + d];
    sq = fmaf(bq[(h << 6) + c], hv, sq);
    sk = fmaf(bk[(h << 6) + c], hv, sk);
  }
  bqE[n] = sq * cp;
  bkE[n] = sk * cp;
}

// ----------------------------------------------------------------------------
// Generic LDS-tiled bf16 WMMA GEMM:  C[M,N] = A[M,K] * B(K,N) + bias[N]
//   A: row-major [m][k], f32 (converted while staging) or bf16.
//   B: stored [n][k] (n-major); staged with async global->LDS b128 copies.
// OUT_MODE: 0 -> bf16 [B,H,S,D]   1 -> bf16 [B,H,D,S]
//           2 -> bf16 ctx [b,q,g*64+d] (slab z=b*16+g)   3 -> f32 row-major
// ----------------------------------------------------------------------------
#define LDSK 40   // LDS pitch in bf16: 80 bytes, keeps 16B-aligned frag loads

template <bool AF32, int OUT_MODE, int BM, int BN, int WM, int WN>
__global__ __launch_bounds__(256) void gemm_bf16(
    const void* __restrict__ Ap, const bf16* __restrict__ Bw,
    const float* __restrict__ bias, void* __restrict__ Cp, int Kdim)
{
  static_assert((BM / WM) * (BN / WN) == 8, "8 waves per workgroup");
  constexpr int WAVES_M = BM / WM;
  constexpr int SUBM = WM / 16, SUBN = WN / 16;

  __shared__ __align__(16) bf16 As[BM][LDSK];
  __shared__ __align__(16) bf16 Bs[BN][LDSK];

  const int tid  = threadIdx.x;
  const int lane = tid & 31, wave = tid >> 5;
  const int half = lane >> 4, lm = lane & 15;
  const int wm = wave % WAVES_M, wn = wave / WAVES_M;
  const int m0 = blockIdx.x * BM, n0 = blockIdx.y * BN;
  const int z  = blockIdx.z;

  const float* Af = (const float*)Ap;
  const bf16*  Ah = (const bf16*)Ap;
  const bf16*  Bb = Bw;
  if constexpr (OUT_MODE == 2) {                  // per-(b,g) slabs
    Af += (size_t)z * (SS * SS);                  // probs slab (f32, d_out)
    Ah += (size_t)z * (SS * SS);
    Bb += (size_t)z * (DD * SS);                  // Vt slab
  }

  v8f acc[SUBM][SUBN] = {};

  for (int kk = 0; kk < Kdim; kk += 32) {
#if USE_ASYNC_LDS
    // ---- B tile via CDNA5 async global->LDS (16B per lane per issue) ----
    for (int i = tid; i < (BN * 32) / 8; i += 256) {
      int e = i * 8;
      int r = e >> 5, c = e & 31;
      async_g2l_b128(&Bs[r][c], &Bb[(size_t)(n0 + r) * Kdim + kk + c]);
    }
#else
    for (int i = tid; i < BN * 32; i += 256) {
      int r = i >> 5, c = i & 31;
      Bs[r][c] = Bb[(size_t)(n0 + r) * Kdim + kk + c];
    }
#endif
    // ---- A tile (convert f32->bf16 while staging if needed) ----
    for (int i = tid; i < BM * 32; i += 256) {
      int r = i >> 5, c = i & 31;
      size_t g = (size_t)(m0 + r) * Kdim + kk + c;
      As[r][c] = AF32 ? f2bf(Af[g]) : Ah[g];
    }
    if (kk + 32 < Kdim) {                         // CDNA5 global_prefetch_b8
      size_t g = (size_t)(m0 + (tid & (BM - 1))) * Kdim + kk + 32;
      __builtin_prefetch(AF32 ? (const void*)&Af[g] : (const void*)&Ah[g], 0, 1);
    }
#if USE_ASYNC_LDS
    wait_async0();
#endif
    __syncthreads();

    // ---- fragments per ISA layout ----
    // A 16x32: lane=row m; e<8 -> k=half*8+e ; e>=8 -> k=16+half*8+(e-8)
    BF16x16 af[SUBM];
    #pragma unroll
    for (int sm = 0; sm < SUBM; ++sm) {
      const bf16* p = &As[wm * WM + sm * 16 + lm][half * 8];
      af[sm].h[0] = *(const v8bf*)(p);
      af[sm].h[1] = *(const v8bf*)(p + 16);
    }
    // B 32x16: lane=col n; e<8 -> k=half*8+e ; e>=8 -> k=16+half*8+(e-8)
    BF16x16 bfr[SUBN];
    #pragma unroll
    for (int sn = 0; sn < SUBN; ++sn) {
      const bf16* p = &Bs[wn * WN + sn * 16 + lm][half * 8];
      bfr[sn].h[0] = *(const v8bf*)(p);
      bfr[sn].h[1] = *(const v8bf*)(p + 16);
    }
    #pragma unroll
    for (int sm = 0; sm < SUBM; ++sm)
      #pragma unroll
      for (int sn = 0; sn < SUBN; ++sn)
        acc[sm][sn] = __builtin_amdgcn_wmma_f32_16x16x32_bf16(
            false, af[sm].v, false, bfr[sn].v, (short)0, acc[sm][sn],
            false, false);
    __syncthreads();
  }

  // ---- epilogue: C/D layout -> row M = r + half*8, col N = lm ----
  #pragma unroll
  for (int sm = 0; sm < SUBM; ++sm)
    #pragma unroll
    for (int sn = 0; sn < SUBN; ++sn)
      #pragma unroll
      for (int r = 0; r < 8; ++r) {
        int m = m0 + wm * WM + sm * 16 + half * 8 + r;
        int n = n0 + wn * WN + sn * 16 + lm;
        float v = acc[sm][sn][r];
        if (bias) v += bias[n];
        if constexpr (OUT_MODE == 0) {            // bf16 [B,H,S,D]
          int bb = m >> 10, s = m & 1023, h = n >> 6, d = n & 63;
          ((bf16*)Cp)[((size_t)((bb << 4) + h) * SS + s) * DD + d] = f2bf(v);
        } else if constexpr (OUT_MODE == 1) {     // bf16 [B,H,D,S]
          int bb = m >> 10, s = m & 1023, h = n >> 6, d = n & 63;
          ((bf16*)Cp)[((size_t)((bb << 4) + h) * DD + d) * SS + s] = f2bf(v);
        } else if constexpr (OUT_MODE == 2) {     // ctx -> [b, q, g*64+d] bf16
          int b = z >> 4, g = z & 15;
          ((bf16*)Cp)[((size_t)(b << 10) + m) * EE + (g << 6) + n] = f2bf(v);
        } else {                                  // f32 row-major
          ((float*)Cp)[(size_t)m * EE + n] = v;
        }
      }
}

// ----------------------------------------------------------------------------
// Pass A (stats): one workgroup per (b*H+h, 16 q-rows). 8 waves split the 1024
// keys (16 WMMAs/wave). Row max and 1/sum(exp) over the full key row via
// shfl_xor within 16-lane groups + LDS cross-wave reduction. Writes only the
// 2 x [B*H*S] f32 stats tables (512KB) -- no probs traffic.
// ----------------------------------------------------------------------------
__global__ __launch_bounds__(256) void attn_stats(
    const bf16* __restrict__ Q, const bf16* __restrict__ Kt,
    float* __restrict__ Mstat, float* __restrict__ Lstat)
{
  const int tid  = threadIdx.x;
  const int lane = tid & 31, wave = tid >> 5;
  const int half = lane >> 4, lm = lane & 15;
  const int bh = blockIdx.y;
  const int q0 = blockIdx.x << 4;
  const bf16* Qb = Q  + (size_t)bh * SS * DD;
  const bf16* Kb = Kt + (size_t)bh * SS * DD;

  BF16x16 af[2];
  #pragma unroll
  for (int kk2 = 0; kk2 < 2; ++kk2) {
    const bf16* qp = Qb + (size_t)(q0 + lm) * DD + kk2 * 32 + half * 8;
    af[kk2].h[0] = *(const v8bf*)(qp);
    af[kk2].h[1] = *(const v8bf*)(qp + 16);
  }

  v8f acc[8] = {};
  #pragma unroll
  for (int t = 0; t < 8; ++t) {
    int key = wave * 128 + t * 16 + lm;
    #pragma unroll
    for (int kk2 = 0; kk2 < 2; ++kk2) {
      BF16x16 bfr;
      const bf16* kp = Kb + (size_t)key * DD + kk2 * 32 + half * 8;
      bfr.h[0] = *(const v8bf*)(kp);
      bfr.h[1] = *(const v8bf*)(kp + 16);
      acc[t] = __builtin_amdgcn_wmma_f32_16x16x32_bf16(
          false, af[kk2].v, false, bfr.v, (short)0, acc[t], false, false);
    }
  }

  __shared__ float red[8][16];
  float rmax[8], gmax[8], rsum[8];
  #pragma unroll
  for (int r = 0; r < 8; ++r) rmax[r] = -3.0e38f;
  #pragma unroll
  for (int t = 0; t < 8; ++t)
    #pragma unroll
    for (int r = 0; r < 8; ++r) {
      float v = acc[t][r] * 0.125f;               // 1/sqrt(64)
      acc[t][r] = v;
      rmax[r] = fmaxf(rmax[r], v);
    }
  #pragma unroll
  for (int off = 1; off < 16; off <<= 1)
    #pragma unroll
    for (int r = 0; r < 8; ++r)
      rmax[r] = fmaxf(rmax[r], __shfl_xor(rmax[r], off, 32));
  if (lm == 0)
    #pragma unroll
    for (int r = 0; r < 8; ++r) red[wave][half * 8 + r] = rmax[r];
  __syncthreads();
  #pragma unroll
  for (int r = 0; r < 8; ++r) {
    float m = red[0][half * 8 + r];
    #pragma unroll
    for (int w = 1; w < 8; ++w) m = fmaxf(m, red[w][half * 8 + r]);
    gmax[r] = m;
    rsum[r] = 0.f;
  }
  __syncthreads();
  #pragma unroll
  for (int t = 0; t < 8; ++t)
    #pragma unroll
    for (int r = 0; r < 8; ++r)
      rsum[r] += __expf(acc[t][r] - gmax[r]);
  #pragma unroll
  for (int off = 1; off < 16; off <<= 1)
    #pragma unroll
    for (int r = 0; r < 8; ++r)
      rsum[r] += __shfl_xor(rsum[r], off, 32);
  if (lm == 0)
    #pragma unroll
    for (int r = 0; r < 8; ++r) red[wave][half * 8 + r] = rsum[r];
  __syncthreads();
  if (wave == 0 && lm == 0) {
    #pragma unroll
    for (int r = 0; r < 8; ++r) {
      int row = half * 8 + r;
      float s = red[0][row];
      #pragma unroll
      for (int w = 1; w < 8; ++w) s += red[w][row];
      Mstat[(size_t)bh * SS + q0 + row] = gmax[r];
      Lstat[(size_t)bh * SS + q0 + row] = 1.0f / s;
    }
  }
}

// ----------------------------------------------------------------------------
// Pass B (fused probs + head-mix): workgroup per (b, 16 q-rows, 128-key block);
// wave w owns a 16-key window. For each head h: 2 WMMAs rebuild the score
// tile, exp/normalize with the stats tables, then accumulate
// mixed[g] += mask[h,g] * p element-wise in C-fragment registers (the
// (lane,vgpr)->(q,k) map is head-invariant). Post-mix probs written ONCE.
// ----------------------------------------------------------------------------
__global__ __launch_bounds__(256) void attn_probs_mix(
    const bf16* __restrict__ Q, const bf16* __restrict__ Kt,
    const float* __restrict__ Mstat, const float* __restrict__ Lstat,
    const float* __restrict__ ent, float* __restrict__ P)
{
  const int tid  = threadIdx.x;
  const int lane = tid & 31, wave = tid >> 5;
  const int half = lane >> 4, lm = lane & 15;
  const int q0 = blockIdx.x << 4;
  const int b  = blockIdx.z;
  const int ky = (blockIdx.y << 7) + wave * 16;   // this wave's 16 keys

  __shared__ float mask[256];
  __shared__ float Mtab[16][16];
  __shared__ float Ltab[16][16];
  mask[tid] = ent[tid];
  {
    int h = tid >> 4, qq = tid & 15;
    size_t sidx = (size_t)(b * 16 + h) * SS + q0 + qq;
    Mtab[h][qq] = Mstat[sidx];
    Ltab[h][qq] = Lstat[sidx];
  }
  __syncthreads();

  v8f mixed[16] = {};                             // 16 g-accumulators (128 VGPRs)
  #pragma unroll 1
  for (int h = 0; h < 16; ++h) {
    const bf16* Qb = Q  + (size_t)(b * 16 + h) * SS * DD;
    const bf16* Kb = Kt + (size_t)(b * 16 + h) * SS * DD;
    v8f acc = {};
    #pragma unroll
    for (int kk2 = 0; kk2 < 2; ++kk2) {
      BF16x16 af, bfr;
      const bf16* qp = Qb + (size_t)(q0 + lm) * DD + kk2 * 32 + half * 8;
      af.h[0] = *(const v8bf*)(qp);
      af.h[1] = *(const v8bf*)(qp + 16);
      const bf16* kp = Kb + (size_t)(ky + lm) * DD + kk2 * 32 + half * 8;
      bfr.h[0] = *(const v8bf*)(kp);
      bfr.h[1] = *(const v8bf*)(kp + 16);
      acc = __builtin_amdgcn_wmma_f32_16x16x32_bf16(
          false, af.v, false, bfr.v, (short)0, acc, false, false);
    }
    float p[8];
    #pragma unroll
    for (int r = 0; r < 8; ++r) {
      int row = half * 8 + r;
      p[r] = __expf(acc[r] * 0.125f - Mtab[h][row]) * Ltab[h][row];
    }
    #pragma unroll
    for (int g = 0; g < 16; ++g) {
      float w = mask[h * 16 + g];
      #pragma unroll
      for (int r = 0; r < 8; ++r)
        mixed[g][r] = fmaf(p[r], w, mixed[g][r]);
    }
  }
  // single write of post-mix probs (f32, d_out)
  #pragma unroll
  for (int g = 0; g < 16; ++g)
    #pragma unroll
    for (int r = 0; r < 8; ++r) {
      int row = half * 8 + r;
      P[((size_t)(b * 16 + g) * SS + q0 + row) * SS + ky + lm] = mixed[g][r];
    }
}

// ----------------------------------------------------------------------------
// Host-side launch
// ----------------------------------------------------------------------------
extern "C" void kernel_launch(void* const* d_in, const int* in_sizes, int n_in,
                              void* d_out, int out_size, void* d_ws, size_t ws_size,
                              hipStream_t stream) {
  (void)in_sizes; (void)n_in; (void)out_size; (void)ws_size;
  const float* query = (const float*)d_in[0];
  const float* key_  = (const float*)d_in[1];
  const float* value = (const float*)d_in[2];
  const float* Wq = (const float*)d_in[3];  const float* bq = (const float*)d_in[4];
  const float* Wk = (const float*)d_in[5];  const float* bk = (const float*)d_in[6];
  const float* Wv = (const float*)d_in[7];  const float* bv = (const float*)d_in[8];
  const float* Wo = (const float*)d_in[9];  const float* bo = (const float*)d_in[10];
  const float* phase = (const float*)d_in[11];
  const float* had   = (const float*)d_in[12];
  const float* ent   = (const float*)d_in[13];

  float* out   = (float*)d_out;                     // [B,S,E] = 4M f32
  float* probs = out + (size_t)4 * SS * EE;         // [B,H,S,S] = 64M f32

  // workspace layout (~41 MB)
  const size_t NEE = (size_t)EE * EE;               // 1M elements
  char* ws = (char*)d_ws;
  bf16*  WqE = (bf16*)ws;
  bf16*  WkE = (bf16*)(ws + 2 * NEE);
  bf16*  WvT = (bf16*)(ws + 4 * NEE);
  bf16*  WoT = (bf16*)(ws + 6 * NEE);
  float* bqE = (float*)(ws + 8 * NEE);
  float* bkE = (float*)(ws + 8 * NEE + 4096);
  bf16*  Qws = (bf16*)(ws + 8 * NEE + 8192);
  bf16*  Kws = Qws + 4 * NEE;                       // B*H*S*D = 4M elems each
  bf16*  Vt  = Kws + 4 * NEE;
  bf16*  Ctx = Vt  + 4 * NEE;
  float* Mstat = (float*)(Ctx + 4 * NEE);           // [B*H*S] f32
  float* Lstat = Mstat + (size_t)4 * HH * SS;       // [B*H*S] f32

  // 1) fold Hadamard + phase into weights, convert to bf16
  prep_weights<<<dim3(4096), 256, 0, stream>>>(Wq, Wk, Wv, Wo, phase, had,
                                               WqE, WkE, WvT, WoT);
  prep_bias<<<dim3(4), 256, 0, stream>>>(bq, bk, phase, had, bqE, bkE);

  // 2-4) projections: M=B*S=4096, N=E=1024, K=E=1024
  gemm_bf16<true, 0, 128, 128, 32, 64><<<dim3(32, 8, 1), 256, 0, stream>>>(
      query, WqE, bqE, Qws, EE);
  gemm_bf16<true, 0, 128, 128, 32, 64><<<dim3(32, 8, 1), 256, 0, stream>>>(
      key_, WkE, bkE, Kws, EE);
  gemm_bf16<true, 1, 128, 128, 32, 64><<<dim3(32, 8, 1), 256, 0, stream>>>(
      value, WvT, bv, Vt, EE);

  // 5) pass A: softmax stats only (512KB)
  attn_stats<<<dim3(64, 64), 256, 0, stream>>>(Qws, Kws, Mstat, Lstat);

  // 6) pass B: recompute scores, fused softmax + head-mix, single probs write
  attn_probs_mix<<<dim3(64, 8, 4), 256, 0, stream>>>(Qws, Kws, Mstat, Lstat,
                                                     ent, probs);

  // 7) ctx = probs @ V per (b,g): M=1024, N=64, K=1024, 64 slabs
  gemm_bf16<true, 2, 128, 64, 16, 64><<<dim3(8, 1, 64), 256, 0, stream>>>(
      probs, Vt, nullptr, Ctx, SS);

  // 8) out = ctx @ Wo.T + bo -> f32 d_out
  gemm_bf16<false, 3, 128, 128, 32, 64><<<dim3(32, 8, 1), 256, 0, stream>>>(
      Ctx, WoT, bo, out, EE);
}
